// Block_21406117003497
// MI455X (gfx1250) — matrix operand order repaired and checked by reading, not compiled, not run on hardware
//
#include <hip/hip_runtime.h>
#include <hip/hip_bf16.h>

#define TSEQ 2048
#define CDIM 1024
#define NHEAD 16
#define HSZ 64
#define DFFD 4096
#define BB 4
#define MROWS (BB * TSEQ)   // 8192

typedef __attribute__((ext_vector_type(16))) __bf16 v16bf;
typedef __attribute__((ext_vector_type(8)))  float  v8f;
typedef __attribute__((ext_vector_type(4)))  float  f4;

union Frag16 { v16bf v; f4 f[2]; };

__device__ __forceinline__ v8f wmma_bf16(v16bf a, v16bf b, v8f c) {
  return __builtin_amdgcn_wmma_f32_16x16x32_bf16(false, a, false, b, (short)0, c,
                                                 false, false);
}

// A fragment: 16x32 bf16, rows m = lane&15, K pattern per ISA 7.12.2:
// lanes 0-15 hold K {0..7,16..23}, lanes 16-31 hold K {8..15,24..31}
__device__ __forceinline__ v16bf load_a_frag(const __bf16* __restrict__ src,
                                             int ld, int row0, int k0) {
  int lane = threadIdx.x & 31;
  int m = lane & 15;
  int hi = lane >> 4;
  const __bf16* p = src + (size_t)(row0 + m) * ld + k0 + hi * 8;
  Frag16 f;
  f.f[0] = *(const f4*)p;
  f.f[1] = *(const f4*)(p + 16);
  return f.v;
}

// B fragment: 32x16 bf16 from [N][K]-major storage; lane holds col n = lane&15,
// lanes 0-15 hold K {0..7,16..23}, lanes 16-31 hold K {8..15,24..31} (matches A).
__device__ __forceinline__ v16bf load_b_frag(const __bf16* __restrict__ wt,
                                             int ld, int n0, int k0) {
  int lane = threadIdx.x & 31;
  int n = lane & 15;
  int hi = lane >> 4;
  const __bf16* p = wt + (size_t)(n0 + n) * ld + k0 + hi * 8;
  Frag16 f;
  f.f[0] = *(const f4*)p;
  f.f[1] = *(const f4*)(p + 16);
  return f.v;
}

// ---------------- weight preprocessing (fp32 -> bf16, transposed) ------------

// Wq/Wk/Wv: [H][C][HS] fp32 -> Wt[(h*HS+d)][c] bf16
__global__ void wt_head_kernel(const float* __restrict__ W, __bf16* __restrict__ Wt) {
  size_t i = (size_t)blockIdx.x * 256 + threadIdx.x;  // H*C*HS = 1048576
  int d = (int)(i & 63);
  size_t r = i >> 6;
  int c = (int)(r & (CDIM - 1));
  int h = (int)(r >> 10);
  Wt[((size_t)(h * HSZ + d) << 10) + c] = (__bf16)W[i];
}

// W: [K][N] fp32 -> Wt[N][K] bf16
__global__ void wt_plain_kernel(const float* __restrict__ W, __bf16* __restrict__ Wt,
                                int Kd, int Nd) {
  size_t i = (size_t)blockIdx.x * 256 + threadIdx.x;
  if (i >= (size_t)Kd * Nd) return;
  int n = (int)(i % Nd);
  int k = (int)(i / Nd);
  Wt[(size_t)n * Kd + k] = (__bf16)W[i];
}

// ---------------- layernorm (fp32 in -> bf16 out) ----------------------------

__global__ void ln_kernel(const float* __restrict__ x, const float* __restrict__ g,
                          const float* __restrict__ bt, __bf16* __restrict__ out) {
  int row = blockIdx.x;
  int tid = threadIdx.x;          // 256 threads, 4 elems each (C=1024)
  const f4* xr = (const f4*)(x + (size_t)row * CDIM);
  f4 v = xr[tid];
  float s = v.x + v.y + v.z + v.w;
#pragma unroll
  for (int m = 1; m < 32; m <<= 1) s += __shfl_xor(s, m, 32);
  __shared__ float sm[8];
  int wave = tid >> 5, lane = tid & 31;
  if (lane == 0) sm[wave] = s;
  __syncthreads();
  float tot = 0.0f;
#pragma unroll
  for (int i = 0; i < 8; ++i) tot += sm[i];
  float mu = tot * (1.0f / CDIM);
  f4 d;
  d.x = v.x - mu; d.y = v.y - mu; d.z = v.z - mu; d.w = v.w - mu;
  float q = d.x * d.x + d.y * d.y + d.z * d.z + d.w * d.w;
#pragma unroll
  for (int m = 1; m < 32; m <<= 1) q += __shfl_xor(q, m, 32);
  __syncthreads();
  if (lane == 0) sm[wave] = q;
  __syncthreads();
  float vt = 0.0f;
#pragma unroll
  for (int i = 0; i < 8; ++i) vt += sm[i];
  float rstd = rsqrtf(vt * (1.0f / CDIM) + 1e-5f);
  const f4 gg = ((const f4*)g)[tid];
  const f4 bb = ((const f4*)bt)[tid];
  __bf16* orow = out + (size_t)row * CDIM + tid * 4;
  orow[0] = (__bf16)(d.x * rstd * gg.x + bb.x);
  orow[1] = (__bf16)(d.y * rstd * gg.y + bb.y);
  orow[2] = (__bf16)(d.z * rstd * gg.z + bb.z);
  orow[3] = (__bf16)(d.w * rstd * gg.w + bb.w);
}

// ---------------- generic WMMA GEMM ------------------------------------------
// C[M,N] = A[M,K](bf16, row-major) x Bt[N,K](bf16) [+bias] [relu] [+resid]
// block: 256 threads = 8 waves; wave tile 32(M) x 64(N) -> block tile 256 x 64.
// Per k-step: 12 b128 loads feed 8 WMMAs (B fragments reused across 2 M-tiles).

template <bool BIAS, bool RELU, bool RESID, bool OUTF, bool OUTB>
__global__ __launch_bounds__(256) void gemm_kernel(
    const __bf16* __restrict__ A, const __bf16* __restrict__ Bt,
    const float* __restrict__ bias, const float* __restrict__ resid,
    float* __restrict__ outf, __bf16* __restrict__ outb, int N, int K) {
  int wave = threadIdx.x >> 5;
  int m0 = blockIdx.x * 256 + wave * 32;
  int n0 = blockIdx.y * 64;
  v8f acc0[4] = {};   // rows m0..m0+15
  v8f acc1[4] = {};   // rows m0+16..m0+31
  {
    int lane = threadIdx.x & 31;
    const __bf16* pf = A + (size_t)(m0 + (lane & 15)) * K;
    for (int k = 0; k < K; k += 32) {
      __builtin_prefetch(pf + k + 64, 0, 0);  // next-next A chunk
      v16bf a0 = load_a_frag(A, K, m0, k);
      v16bf a1 = load_a_frag(A, K, m0 + 16, k);
#pragma unroll
      for (int j = 0; j < 4; ++j) {
        v16bf b = load_b_frag(Bt, K, n0 + 16 * j, k);
        acc0[j] = wmma_bf16(a0, b, acc0[j]);
        acc1[j] = wmma_bf16(a1, b, acc1[j]);
      }
    }
  }
  int lane = threadIdx.x & 31;
  int nn = lane & 15, hi = lane >> 4;
#pragma unroll
  for (int j = 0; j < 4; ++j) {
    int col = n0 + 16 * j + nn;
    float bv = BIAS ? bias[col] : 0.0f;
#pragma unroll
    for (int e = 0; e < 8; ++e) {
#pragma unroll
      for (int half = 0; half < 2; ++half) {
        int row = m0 + half * 16 + e + 8 * hi;
        float v = (half ? acc1[j][e] : acc0[j][e]) + bv;
        if (RELU) v = fmaxf(v, 0.0f);
        if (RESID) v += resid[(size_t)row * N + col];
        if (OUTF) outf[(size_t)row * N + col] = v;
        if (OUTB) outb[(size_t)row * N + col] = (__bf16)v;
      }
    }
  }
}

// ---------------- V transpose: [B*T, C] -> per (b,h): [HS][T] ----------------

__global__ void transpose_v_kernel(const __bf16* __restrict__ V,
                                   __bf16* __restrict__ Vt) {
  size_t i = (size_t)blockIdx.x * 256 + threadIdx.x;  // B*T*C = 8388608
  int col = (int)(i & (CDIM - 1));
  size_t bt = i >> 10;
  int t = (int)(bt & (TSEQ - 1));
  int b = (int)(bt >> 11);
  int h = col >> 6, d = col & 63;
  Vt[(((size_t)(b * NHEAD + h) * HSZ + d) << 11) + t] = V[i];
}

// ---------------- flash attention --------------------------------------------
// grid (T/64, B*H), 128 threads = 4 waves; each wave: 16 query rows x full d=64.

__global__ __launch_bounds__(128) void attn_kernel(
    const __bf16* __restrict__ Q, const __bf16* __restrict__ Km,
    const __bf16* __restrict__ Vt, __bf16* __restrict__ O) {
  int bh = blockIdx.y;
  int b = bh >> 4;
  int h = bh & 15;
  int wave = threadIdx.x >> 5;
  int lane = threadIdx.x & 31;
  int nn = lane & 15, hi = lane >> 4;
  int q0 = blockIdx.x * 64 + wave * 16;

  const __bf16* Qb = Q + (size_t)b * TSEQ * CDIM + h * HSZ;
  const __bf16* Kb = Km + (size_t)b * TSEQ * CDIM + h * HSZ;
  const __bf16* Vb = Vt + (size_t)(b * NHEAD + h) * HSZ * TSEQ;  // [64][T]

  v16bf qa0 = load_a_frag(Qb, CDIM, q0, 0);
  v16bf qa1 = load_a_frag(Qb, CDIM, q0, 32);

  v8f o0 = {}, o1 = {}, o2 = {}, o3 = {};
  float rmax[8], rsum[8];
#pragma unroll
  for (int e = 0; e < 8; ++e) { rmax[e] = -3.0e38f; rsum[e] = 0.0f; }

  __shared__ __align__(16) __bf16 pbuf[4][16 * 32];  // per-wave P staging
  __bf16* pw = pbuf[wave];

  const float scale = 0.03125f;  // C**-0.5 = 1/sqrt(1024)

  int kend = q0 + 16;
  for (int kb = 0; kb < kend; kb += 32) {
    // S = Q * K^T for 32 keys (two 16x16 tiles), d=64 -> two k-steps each
    v8f s0 = {}, s1 = {};
    s0 = wmma_bf16(qa0, load_b_frag(Kb, CDIM, kb, 0), s0);
    s0 = wmma_bf16(qa1, load_b_frag(Kb, CDIM, kb, 32), s0);
    s1 = wmma_bf16(qa0, load_b_frag(Kb, CDIM, kb + 16, 0), s1);
    s1 = wmma_bf16(qa1, load_b_frag(Kb, CDIM, kb + 16, 32), s1);

#pragma unroll
    for (int e = 0; e < 8; ++e) {
      int qi = q0 + e + 8 * hi;
      float v0 = (kb + nn <= qi) ? s0[e] * scale : -3.0e38f;
      float v1 = (kb + 16 + nn <= qi) ? s1[e] * scale : -3.0e38f;
      float mx = fmaxf(v0, v1);
#pragma unroll
      for (int md = 1; md < 16; md <<= 1) mx = fmaxf(mx, __shfl_xor(mx, md, 32));
      float nmax = fmaxf(rmax[e], mx);
      float corr = __expf(rmax[e] - nmax);
      rmax[e] = nmax;
      float p0 = __expf(v0 - nmax);
      float p1 = __expf(v1 - nmax);
      float ps = p0 + p1;
#pragma unroll
      for (int md = 1; md < 16; md <<= 1) ps += __shfl_xor(ps, md, 32);
      rsum[e] = rsum[e] * corr + ps;
      o0[e] *= corr; o1[e] *= corr; o2[e] *= corr; o3[e] *= corr;
      int prow = e + 8 * hi;
      pw[prow * 32 + nn] = (__bf16)p0;
      pw[prow * 32 + 16 + nn] = (__bf16)p1;
    }
    // O += P * V (P restaged via LDS into A-fragment layout)
    v16bf pa = load_a_frag(pw, 32, 0, 0);
    o0 = wmma_bf16(pa, load_b_frag(Vb, TSEQ, 0, kb), o0);
    o1 = wmma_bf16(pa, load_b_frag(Vb, TSEQ, 16, kb), o1);
    o2 = wmma_bf16(pa, load_b_frag(Vb, TSEQ, 32, kb), o2);
    o3 = wmma_bf16(pa, load_b_frag(Vb, TSEQ, 48, kb), o3);
  }

  __bf16* Ob = O + (size_t)b * TSEQ * CDIM + h * HSZ;
#pragma unroll
  for (int e = 0; e < 8; ++e) {
    float inv = 1.0f / rsum[e];
    int t = q0 + e + 8 * hi;
    __bf16* orow = Ob + (size_t)t * CDIM;
    orow[nn]      = (__bf16)(o0[e] * inv);
    orow[16 + nn] = (__bf16)(o1[e] * inv);
    orow[32 + nn] = (__bf16)(o2[e] * inv);
    orow[48 + nn] = (__bf16)(o3[e] * inv);
  }
}

// ---------------- launch -----------------------------------------------------

extern "C" void kernel_launch(void* const* d_in, const int* in_sizes, int n_in,
                              void* d_out, int out_size, void* d_ws, size_t ws_size,
                              hipStream_t stream) {
  const float* x     = (const float*)d_in[0];
  const float* Wq    = (const float*)d_in[1];
  const float* Wk    = (const float*)d_in[2];
  const float* Wv    = (const float*)d_in[3];
  const float* Wproj = (const float*)d_in[4];
  const float* bproj = (const float*)d_in[5];
  const float* ln1g  = (const float*)d_in[6];
  const float* ln1b  = (const float*)d_in[7];
  const float* ln2g  = (const float*)d_in[8];
  const float* ln2b  = (const float*)d_in[9];
  const float* W1    = (const float*)d_in[10];
  const float* b1    = (const float*)d_in[11];
  const float* W2    = (const float*)d_in[12];
  const float* b2    = (const float*)d_in[13];
  float* out = (float*)d_out;
  (void)in_sizes; (void)n_in; (void)out_size; (void)ws_size;

  char* ws = (char*)d_ws;
  size_t off = 0;
  auto alloc = [&](size_t bytes) -> char* {
    char* p = ws + off;
    off += (bytes + 255) & ~(size_t)255;
    return p;
  };
  const size_t szC2 = (size_t)CDIM * CDIM;   // 1M
  const size_t szCF = (size_t)CDIM * DFFD;   // 4M
  const size_t szA  = (size_t)MROWS * CDIM;  // 8M

  __bf16* wq_t = (__bf16*)alloc(szC2 * 2);
  __bf16* wk_t = (__bf16*)alloc(szC2 * 2);
  __bf16* wv_t = (__bf16*)alloc(szC2 * 2);
  __bf16* wp_t = (__bf16*)alloc(szC2 * 2);
  __bf16* w1_t = (__bf16*)alloc(szCF * 2);
  __bf16* w2_t = (__bf16*)alloc(szCF * 2);
  __bf16* hbuf = (__bf16*)alloc(szA * 2);    // ln1 output, later ln2 output
  float*  x2   = (float*)alloc(szA * 4);     // post-attention residual state
  __bf16* obuf = (__bf16*)alloc(szA * 2);    // attention output (concat heads)
  __bf16* qbuf = (__bf16*)alloc(szA * 2);
  __bf16* kbuf = (__bf16*)alloc(szA * 2);
  __bf16* vbuf = (__bf16*)alloc(szA * 2);
  __bf16* vtbf = (__bf16*)alloc(szA * 2);
  __bf16* ffbf = qbuf;  // ff (64MB) aliases q/k/v/vt block (done before MLP)

  // --- weight preprocessing ---
  wt_head_kernel<<<4096, 256, 0, stream>>>(Wq, wq_t);
  wt_head_kernel<<<4096, 256, 0, stream>>>(Wk, wk_t);
  wt_head_kernel<<<4096, 256, 0, stream>>>(Wv, wv_t);
  wt_plain_kernel<<<4096, 256, 0, stream>>>(Wproj, wp_t, CDIM, CDIM);
  wt_plain_kernel<<<16384, 256, 0, stream>>>(W1, w1_t, CDIM, DFFD);
  wt_plain_kernel<<<16384, 256, 0, stream>>>(W2, w2_t, DFFD, CDIM);

  // --- ln1 ---
  ln_kernel<<<MROWS, 256, 0, stream>>>(x, ln1g, ln1b, hbuf);

  // --- Q, K, V projections ---
  dim3 gQ(MROWS / 256, CDIM / 64);
  gemm_kernel<false, false, false, false, true>
      <<<gQ, 256, 0, stream>>>(hbuf, wq_t, nullptr, nullptr, nullptr, qbuf, CDIM, CDIM);
  gemm_kernel<false, false, false, false, true>
      <<<gQ, 256, 0, stream>>>(hbuf, wk_t, nullptr, nullptr, nullptr, kbuf, CDIM, CDIM);
  gemm_kernel<false, false, false, false, true>
      <<<gQ, 256, 0, stream>>>(hbuf, wv_t, nullptr, nullptr, nullptr, vbuf, CDIM, CDIM);

  // --- V transpose per head ---
  transpose_v_kernel<<<32768, 256, 0, stream>>>(vbuf, vtbf);

  // --- flash attention ---
  attn_kernel<<<dim3(TSEQ / 64, BB * NHEAD), 128, 0, stream>>>(qbuf, kbuf, vtbf, obuf);

  // --- output projection + bias + residual -> x2 (fp32) ---
  gemm_kernel<true, false, true, true, false>
      <<<gQ, 256, 0, stream>>>(obuf, wp_t, bproj, x, x2, nullptr, CDIM, CDIM);

  // --- ln2 ---
  ln_kernel<<<MROWS, 256, 0, stream>>>(x2, ln2g, ln2b, hbuf);

  // --- MLP layer 1: relu(h2 @ W1 + b1) -> bf16 ---
  dim3 gF(MROWS / 256, DFFD / 64);
  gemm_kernel<true, true, false, false, true>
      <<<gF, 256, 0, stream>>>(hbuf, w1_t, b1, nullptr, nullptr, ffbf, DFFD, CDIM);

  // --- MLP layer 2: ff @ W2 + b2 + x2 -> out (fp32) ---
  gemm_kernel<true, false, true, true, false>
      <<<gQ, 256, 0, stream>>>(ffbf, w2_t, b2, x2, out, nullptr, CDIM, DFFD);
}
